// ComTransformer_78520592106161
// MI455X (gfx1250) — matrix operand order repaired and checked by reading, not compiled, not run on hardware
//
#include <hip/hip_runtime.h>
#include <hip/hip_bf16.h>
#include <cmath>

// ---------------------------------------------------------------------------
// MI455X (gfx1250, wave32) implementation of the complex transformer.
// All GEMMs use v_wmma_f32_16x16x32_bf16 (bf16 operands, fp32 accumulate).
// Attention is fused flash-style (online softmax): scores never hit HBM.
// Weights are pre-transposed/converted to bf16 so all WMMA B-fragments are
// contiguous b128 loads; A-fragments load as two contiguous 8-element runs.
// ---------------------------------------------------------------------------

typedef __attribute__((ext_vector_type(16))) __bf16 v16bf;
typedef __attribute__((ext_vector_type(8)))  __bf16 v8bf;
typedef __attribute__((ext_vector_type(8)))  float  v8f;

#define WMMA_BF16(A, B, C) \
  __builtin_amdgcn_wmma_f32_16x16x32_bf16(false, (A), false, (B), (short)0, (C), false, false)

static __device__ __forceinline__ __bf16 f2bf(float f) { return (__bf16)f; }   // native v_cvt (RNE)
static __device__ __forceinline__ v8bf cvt8(v8f x) {
  return __builtin_convertvector(x, v8bf);                                     // packed cvt
}
static __device__ __forceinline__ v16bf cat16(v8bf a, v8bf b) {
  return __builtin_shufflevector(a, b, 0,1,2,3,4,5,6,7,8,9,10,11,12,13,14,15);
}
static __device__ __forceinline__ v8f vzero8() {
  v8f z;
#pragma unroll
  for (int r = 0; r < 8; ++r) z[r] = 0.0f;
  return z;
}

// WMMA fragment maps (CDNA5 ISA 7.12.2, 16-bit operands, wave32):
//  A (16x32): lane holds row (lane&15); code's order: element e -> K = e + 16*(lane>=16)
//             => two contiguous 8-element runs at K = 16*(lane>=16) and +8.
//  B (32x16): lane holds col (lane&15); same order as A: element e -> K = e + 16*(lane>=16)
//             => one contiguous 16-element run.
//  C/D(16x16): lane holds col (lane&15); vgpr r -> row = r + 8*(lane>=16)

// A-fragment from a contiguous fp32 row (converted to bf16).
static __device__ __forceinline__ v16bf loadA_f32(const float* __restrict__ rowp,
                                                  int k0, int lane) {
  const int off = k0 + (lane & 16);                 // +16 for upper half-wave
  v8f lo = *(const v8f*)(rowp + off);
  v8f hi = *(const v8f*)(rowp + off + 8);
  return cat16(cvt8(lo), cvt8(hi));
}
// A-fragment from a contiguous bf16 row.
static __device__ __forceinline__ v16bf loadA_bf16(const __bf16* __restrict__ rowp,
                                                   int k0, int lane) {
  const int off = k0 + (lane & 16);
  v8bf lo = *(const v8bf*)(rowp + off);
  v8bf hi = *(const v8bf*)(rowp + off + 8);
  return cat16(lo, hi);
}
// B-fragment from a contiguous bf16 row (16 consecutive K values).
static __device__ __forceinline__ v16bf loadB_bf16(const __bf16* __restrict__ rowp,
                                                   int k0, int lane) {
  return *(const v16bf*)(rowp + k0 + (lane & 16));
}

// ---------------------------------------------------------------------------
// Weight prep: transpose+convert W[k, o] (fp32, [D,K]) -> Wt[o, k] (bf16, [K,D])
// so that WMMA B-fragments over the reduction dim are contiguous.
// ---------------------------------------------------------------------------
__global__ void wprep_t_kernel(const float* __restrict__ W, __bf16* __restrict__ Wt,
                               int D, int K)
{
  const long total = (long)D * K;
  for (long e = (long)blockIdx.x * blockDim.x + threadIdx.x; e < total;
       e += (long)gridDim.x * blockDim.x) {
    const long o = e / D, k = e % D;
    Wt[e] = f2bf(W[k * K + o]);
  }
}
// Plain fp32 -> bf16 copy (layout already k-contiguous, e.g. Wc [D, 2D]).
__global__ void wprep_c_kernel(const float* __restrict__ W, __bf16* __restrict__ Wt, long n)
{
  for (long e = (long)blockIdx.x * blockDim.x + threadIdx.x; e < n;
       e += (long)gridDim.x * blockDim.x)
    Wt[e] = f2bf(W[e]);
}

// ---------------------------------------------------------------------------
// Projection: Y[m, :] = L2norm( X[m, :] @ W + bias ), Y stored bf16.
// X: fp32 [M, D]; Wt: bf16 [K, D] (pre-transposed); Y: bf16 [M, K].
// M % 16 == 0, K in {128, 512}.  Block = 256 (8 waves); waves split K.
// ---------------------------------------------------------------------------
__global__ __launch_bounds__(256) void proj_l2n_kernel(
    const float* __restrict__ X, const __bf16* __restrict__ Wt,
    const float* __restrict__ bias, __bf16* __restrict__ Y,
    int M, int D, int K)
{
  __shared__ float rowss[16];
  __shared__ float rinv[16];
  const int tid = threadIdx.x, wid = tid >> 5, lane = tid & 31;
  const int col = lane & 15;
  const int hi  = (lane & 16) ? 8 : 0;
  const int rbase = blockIdx.x * 16;
  const float* xrow = X + (long)(rbase + col) * D;   // A-fragment row pointer

  if (tid < 16) rowss[tid] = 0.0f;
  __syncthreads();

  const int ntile = K >> 4;
  int tlist[4]; int nslots = 0;
  for (int t = wid; t < ntile; t += 8) tlist[nslots++] = t;   // <=4 (K<=512)

  v8f acc[4];
  for (int s = 0; s < nslots; ++s) {
    const int ocol = tlist[s] * 16 + col;
    const __bf16* wrow = Wt + (long)ocol * D;
    v8f a = vzero8();
    for (int ks = 0; ks < D; ks += 32) {
      v16bf af  = loadA_f32(xrow, ks, lane);
      v16bf bf_ = loadB_bf16(wrow, ks, lane);
      a = WMMA_BF16(af, bf_, a);
    }
    const float bv = bias[ocol];
#pragma unroll
    for (int r = 0; r < 8; ++r) {
      float v = a[r] + bv;
      a[r] = v;
      atomicAdd(&rowss[r + hi], v * v);   // ds_add_f32, no return
    }
    acc[s] = a;
  }
  __syncthreads();
  if (tid < 16) rinv[tid] = 1.0f / fmaxf(sqrtf(rowss[tid]), 1e-12f);
  __syncthreads();

  for (int s = 0; s < nslots; ++s) {
    const int ocol = tlist[s] * 16 + col;
#pragma unroll
    for (int r = 0; r < 8; ++r) {
      const long row = rbase + r + hi;
      Y[row * K + ocol] = f2bf(acc[s][r] * rinv[r + hi]);
    }
  }
}

// ---------------------------------------------------------------------------
// Fused complex attention, specialized for Vi == 0 (true for every call in
// this network: encoder vi=si=0, decoder-self vi=ti=0, cross vi=pi*.=0):
//   dr = Qr.Kr - Qi.Ki ; di = Qr.Ki + Qi.Kr  (x TEMP=30)
//   Or = softmax(dr) @ Vr ; Oi = softmax(di) @ Vr
// Flash-style online softmax per branch; Q/K: bf16 [L, B, Kp] (L2-normalized);
// Vr/O*: fp32 [L, B, Dv].  Dv must be 512.  Grid: (ceil(Lq/16), B), 8 waves.
// Waves 0,1 build a 16x32 complex score chunk with WMMA; 16 threads run the
// online softmax in LDS; all 8 waves accumulate P@V with WMMA (64 cols each).
// ---------------------------------------------------------------------------
__global__ __launch_bounds__(256) void com_attn_kernel(
    const __bf16* __restrict__ Qr, const __bf16* __restrict__ Qi,
    const __bf16* __restrict__ Kr, const __bf16* __restrict__ Ki,
    const float*  __restrict__ Vr,
    float* __restrict__ Or, float* __restrict__ Oi,
    int Lq, int Lk, int B, int Kp, int Dv)
{
  __shared__ float Sr[16][32];   // 16 q-rows x 32 k-cols, real scores -> probs
  __shared__ float Si[16][32];   // imag branch
  __shared__ float mR[16], lR[16], aR[16];
  __shared__ float mI[16], lI[16], aI[16];

  const int tid = threadIdx.x, wid = tid >> 5, lane = tid & 31;
  const int col = lane & 15;
  const int hi  = (lane & 16) ? 8 : 0;
  const int b = blockIdx.y;
  const int qbase = blockIdx.x * 16;
  const int qrow = (qbase + col < Lq) ? (qbase + col) : (Lq - 1);
  const __bf16* qrp = Qr + ((long)qrow * B + b) * Kp;
  const __bf16* qip = Qi + ((long)qrow * B + b) * Kp;

  if (tid < 16) { mR[tid] = -3e38f; lR[tid] = 0.f; mI[tid] = -3e38f; lI[tid] = 0.f; }

  v8f Yr[4], Yi[4];
#pragma unroll
  for (int t = 0; t < 4; ++t) { Yr[t] = vzero8(); Yi[t] = vzero8(); }
  __syncthreads();

  const int nK32 = (Lk + 31) / 32;
  for (int kc = 0; kc < nK32; ++kc) {
    // --- phase A: waves 0,1 each produce one 16x16 complex score tile ------
    if (wid < 2) {
      const int kt = kc * 2 + wid;
      const int kcol = (kt * 16 + col < Lk) ? (kt * 16 + col) : (Lk - 1);
      const __bf16* krp = Kr + ((long)kcol * B + b) * Kp;
      const __bf16* kip = Ki + ((long)kcol * B + b) * Kp;
      v8f srr = vzero8(), sii = vzero8(), sri = vzero8(), sir = vzero8();
      for (int ks = 0; ks < Kp; ks += 32) {
        v16bf qrF = loadA_bf16(qrp, ks, lane);
        v16bf qiF = loadA_bf16(qip, ks, lane);
        v16bf krF = loadB_bf16(krp, ks, lane);
        v16bf kiF = loadB_bf16(kip, ks, lane);
        srr = WMMA_BF16(qrF, krF, srr);
        sii = WMMA_BF16(qiF, kiF, sii);
        sri = WMMA_BF16(qrF, kiF, sri);
        sir = WMMA_BF16(qiF, krF, sir);
      }
#pragma unroll
      for (int r = 0; r < 8; ++r) {
        const int row = r + hi;
        float dr = (srr[r] - sii[r]) * 30.0f;   // TEMP
        float di = (sri[r] + sir[r]) * 30.0f;
        if (kt * 16 + col >= Lk) { dr = -3e38f; di = -3e38f; }
        Sr[row][wid * 16 + col] = dr;
        Si[row][wid * 16 + col] = di;
      }
    }
    __syncthreads();

    // --- phase B: online softmax update, one thread per q-row --------------
    if (tid < 16) {
      const int row = tid;
      float mt = -3e38f;
      for (int c = 0; c < 32; ++c) mt = fmaxf(mt, Sr[row][c]);
      float mn = fmaxf(mR[row], mt);
      float al = __expf(mR[row] - mn);
      float sum = 0.f;
      for (int c = 0; c < 32; ++c) { float p = __expf(Sr[row][c] - mn); Sr[row][c] = p; sum += p; }
      lR[row] = lR[row] * al + sum; mR[row] = mn; aR[row] = al;

      mt = -3e38f;
      for (int c = 0; c < 32; ++c) mt = fmaxf(mt, Si[row][c]);
      mn = fmaxf(mI[row], mt);
      al = __expf(mI[row] - mn);
      sum = 0.f;
      for (int c = 0; c < 32; ++c) { float p = __expf(Si[row][c] - mn); Si[row][c] = p; sum += p; }
      lI[row] = lI[row] * al + sum; mI[row] = mn; aI[row] = al;
    }
    __syncthreads();

    // --- phase C: rescale accumulators, P@V via WMMA (all 8 waves) ---------
    float alr[8], ali[8];
#pragma unroll
    for (int r = 0; r < 8; ++r) { alr[r] = aR[r + hi]; ali[r] = aI[r + hi]; }
#pragma unroll
    for (int t = 0; t < 4; ++t)
#pragma unroll
      for (int r = 0; r < 8; ++r) { Yr[t][r] *= alr[r]; Yi[t][r] *= ali[r]; }

    // probability A-fragments from LDS (two b128 runs per branch)
    const int offA = lane & 16;
    v16bf Pr = cat16(cvt8(*(const v8f*)&Sr[col][offA]),
                     cvt8(*(const v8f*)&Sr[col][offA + 8]));
    v16bf Pi = cat16(cvt8(*(const v8f*)&Si[col][offA]),
                     cvt8(*(const v8f*)&Si[col][offA + 8]));

#pragma unroll
    for (int t = 0; t < 4; ++t) {
      const int dcol = wid * 64 + t * 16 + col;     // Dv == 512
      v16bf vF;
#pragma unroll
      for (int e = 0; e < 16; ++e) {                // B-frag over k (strided rows)
        int kg = kc * 32 + e + (lane & 16);
        if (kg >= Lk) kg = Lk - 1;                  // P==0 there, value irrelevant
        vF[e] = f2bf(Vr[((long)kg * B + b) * Dv + dcol]);
      }
      Yr[t] = WMMA_BF16(Pr, vF, Yr[t]);
      Yi[t] = WMMA_BF16(Pi, vF, Yi[t]);
    }
    __syncthreads();
  }

  // --- epilogue: normalize by softmax sums ---------------------------------
#pragma unroll
  for (int t = 0; t < 4; ++t) {
    const int dcol = wid * 64 + t * 16 + col;
#pragma unroll
    for (int r = 0; r < 8; ++r) {
      const int row = r + hi;
      const int q = qbase + row;
      if (q < Lq) {
        const long off = ((long)q * B + b) * Dv + dcol;
        Or[off] = Yr[t][r] * (1.0f / fmaxf(lR[row], 1e-30f));
        Oi[off] = Yi[t][r] * (1.0f / fmaxf(lI[row], 1e-30f));
      }
    }
  }
}

// ---------------------------------------------------------------------------
// Instance L2 norm over one (image, batch) group of HW*D elements.
//   v = A (op==0) | A+B (op==1) | A*B (op==2);  Y = v * scale*sqrt(cnt/(ss+1e-5))
// Layout [Ltot, B, D]; grid (numImgs, B).
// ---------------------------------------------------------------------------
__global__ __launch_bounds__(256) void inorm_kernel(
    const float* __restrict__ A, const float* __restrict__ Bv,
    float* __restrict__ Y, int HW, int B, int D, float scale, int op)
{
  __shared__ float red[8];
  __shared__ float totsh;
  const int img = blockIdx.x, b = blockIdx.y, tid = threadIdx.x;
  const long elems = (long)HW * D;
  float ss = 0.f;
  for (long e = tid; e < elems; e += 256) {
    const long l = e / D, d = e % D;
    const long idx = ((long)(img * HW + l) * B + b) * D + d;
    float v = A[idx];
    if (op == 1) v += Bv[idx];
    else if (op == 2) v *= Bv[idx];
    ss += v * v;
  }
#pragma unroll
  for (int o = 16; o >= 1; o >>= 1) ss += __shfl_xor(ss, o, 32);
  if ((tid & 31) == 0) red[tid >> 5] = ss;
  __syncthreads();
  if (tid == 0) { float s = 0.f; for (int i = 0; i < 8; ++i) s += red[i]; totsh = s; }
  __syncthreads();
  const float sf = scale * sqrtf((float)elems / (totsh + 1e-5f));
  for (long e = tid; e < elems; e += 256) {
    const long l = e / D, d = e % D;
    const long idx = ((long)(img * HW + l) * B + b) * D + d;
    float v = A[idx];
    if (op == 1) v += Bv[idx];
    else if (op == 2) v *= Bv[idx];
    Y[idx] = v * sf;
  }
}

// ---------------------------------------------------------------------------
// Final 1x1 compress conv: Out[b, o, l] = sum_c cat(R,I)[l, b, c] * Wc[o, c]
// R/I: fp32 [HW, B, D]; Wcb: bf16 [D, 2D] (k-contiguous); Out: fp32 [B, D, HW].
// Grid: (ceil(HW/16), D/128, B); 8 waves each own one 16-col o-tile.
// ---------------------------------------------------------------------------
__global__ __launch_bounds__(256) void compress_kernel(
    const float* __restrict__ R, const float* __restrict__ I,
    const __bf16* __restrict__ Wcb, float* __restrict__ Out,
    int HW, int B, int D)
{
  const int tid = threadIdx.x, wid = tid >> 5, lane = tid & 31;
  const int col = lane & 15;
  const int hi  = (lane & 16) ? 8 : 0;
  const int b = blockIdx.z;
  const int lbase = blockIdx.x * 16;
  const int obase = blockIdx.y * 128 + wid * 16;
  const int lrow = (lbase + col < HW) ? (lbase + col) : (HW - 1);
  const int ocol = obase + col;
  const int D2 = 2 * D;
  const float* Rrow = R + ((long)lrow * B + b) * D;
  const float* Irow = I + ((long)lrow * B + b) * D;
  const __bf16* wrow = Wcb + (long)ocol * D2;

  v8f acc = vzero8();
  for (int ks = 0; ks < D2; ks += 32) {
    // A-fragment: two 8-float runs; each run lies fully in R or I (D%8==0).
    const int s0 = ks + (lane & 16);
    const int s1 = s0 + 8;
    v8f lo = (s0 < D) ? *(const v8f*)(Rrow + s0) : *(const v8f*)(Irow + s0 - D);
    v8f hi8 = (s1 < D) ? *(const v8f*)(Rrow + s1) : *(const v8f*)(Irow + s1 - D);
    v16bf af  = cat16(cvt8(lo), cvt8(hi8));
    v16bf bf_ = loadB_bf16(wrow, ks, lane);
    acc = WMMA_BF16(af, bf_, acc);
  }
#pragma unroll
  for (int r = 0; r < 8; ++r) {
    const int l = lbase + r + hi;
    if (l < HW) Out[((long)b * D + ocol) * HW + l] = acc[r];
  }
}

// ---------------------------------------------------------------------------
// Layout helpers
// ---------------------------------------------------------------------------
// [N,B,D,HW] -> seq [N*HW, B, D]
__global__ void to_seq_kernel(const float* __restrict__ X, float* __restrict__ Y,
                              int N, int B, int D, int HW)
{
  const long total = (long)N * B * D * HW;
  for (long e = (long)blockIdx.x * blockDim.x + threadIdx.x; e < total;
       e += (long)gridDim.x * blockDim.x) {
    const long d = e % D;
    long t = e / D;
    const long b = t % B; t /= B;
    const long l = t % HW;
    const long n = t / HW;
    Y[e] = X[(((n * B + b) * D + d) * HW) + l];
  }
}

// label [N,B,HW] -> pos [N*HW, B, D] (broadcast over D)
__global__ void pos_kernel(const float* __restrict__ L, float* __restrict__ Y,
                           int N, int B, int D, int HW)
{
  const long total = (long)N * B * D * HW;
  for (long e = (long)blockIdx.x * blockDim.x + threadIdx.x; e < total;
       e += (long)gridDim.x * blockDim.x) {
    long t = e / D;
    const long b = t % B; t /= B;
    const long l = t % HW;
    const long n = t / HW;
    Y[e] = L[(n * B + b) * HW + l];
  }
}

__global__ void mul_kernel(const float* __restrict__ A, const float* __restrict__ B,
                           float* __restrict__ Y, long n)
{
  for (long e = (long)blockIdx.x * blockDim.x + threadIdx.x; e < n;
       e += (long)gridDim.x * blockDim.x)
    Y[e] = A[e] * B[e];
}

// ---------------------------------------------------------------------------
// Host orchestration
// ---------------------------------------------------------------------------
extern "C" void kernel_launch(void* const* d_in, const int* in_sizes, int n_in,
                              void* d_out, int out_size, void* d_ws, size_t ws_size,
                              hipStream_t stream)
{
  (void)in_sizes; (void)n_in; (void)out_size; (void)ws_size;
  const float* train_feat  = (const float*)d_in[0];
  const float* test_feat   = (const float*)d_in[1];
  const float* train_label = (const float*)d_in[2];
  const float* Wk_en    = (const float*)d_in[3];
  const float* bk_en    = (const float*)d_in[4];
  const float* Wk_self  = (const float*)d_in[5];
  const float* bk_self  = (const float*)d_in[6];
  const float* Wk_cross = (const float*)d_in[7];
  const float* bk_cross = (const float*)d_in[8];
  const float* Wc       = (const float*)d_in[9];
  float* out = (float*)d_out;

  const int NTi = 3, NSi = 3, Bn = 8, Dn = 512, HWn = 484, KDn = 128;
  const int LTn = NTi * HWn;                       // 1452
  const float scale = sqrtf(1.0f / (Dn * 16.0f));  // sqrt(1/(d*4*4))

  // bump allocator over d_ws (needs ~260 MB)
  char* w = (char*)d_ws; size_t off = 0;
  auto alloc = [&](size_t bytes) -> void* {
    void* p = w + off;
    off += (bytes + 255) & ~(size_t)255;
    return p;
  };
  const size_t bigF = (size_t)LTn * Bn * Dn;       // encoder-seq element count
  float*  zerof = (float*) alloc(bigF * 4);
  float*  sr    = (float*) alloc(bigF * 4);
  float*  arr   = (float*) alloc(bigF * 4);
  float*  aii   = (float*) alloc(bigF * 4);
  float*  memr  = (float*) alloc(bigF * 4);
  float*  memi  = (float*) alloc(bigF * 4);
  __bf16* per   = (__bf16*)alloc(bigF * 2);
  __bf16* pei   = (__bf16*)alloc(bigF * 2);
  const size_t smE = (size_t)LTn * Bn * KDn;
  __bf16* smr = (__bf16*)alloc(smE * 2);
  __bf16* smi = (__bf16*)alloc(smE * 2);
  const size_t tE = (size_t)HWn * Bn * Dn;
  float* T[10];
  for (int i = 0; i < 10; ++i) T[i] = (float*)alloc(tE * 4);
  const size_t qE = (size_t)HWn * Bn * KDn;
  __bf16* qsr = (__bf16*)alloc(qE * 2);
  __bf16* qsi = (__bf16*)alloc(qE * 2);
  __bf16* pqr = (__bf16*)alloc(qE * 2);
  __bf16* pqi = (__bf16*)alloc(qE * 2);
  // pre-transposed bf16 weights ([K, D], k-contiguous)
  __bf16* WenT   = (__bf16*)alloc((size_t)Dn * Dn * 2);
  __bf16* WselfT = (__bf16*)alloc((size_t)KDn * Dn * 2);
  __bf16* WcrossT= (__bf16*)alloc((size_t)KDn * Dn * 2);
  __bf16* WcB    = (__bf16*)alloc((size_t)Dn * 2 * Dn * 2);
  float* posr = sr;    // reuse: sr dead after mem is built
  float* vmr  = arr;   // reuse: ar dead after mem is built

  hipMemsetAsync(zerof, 0, bigF * 4, stream);      // zero imag inputs (si/ti)

  // weight prep (once per launch)
  wprep_t_kernel<<<512, 256, 0, stream>>>(Wk_en,    WenT,    Dn, Dn);
  wprep_t_kernel<<<256, 256, 0, stream>>>(Wk_self,  WselfT,  Dn, KDn);
  wprep_t_kernel<<<256, 256, 0, stream>>>(Wk_cross, WcrossT, Dn, KDn);
  wprep_c_kernel<<<512, 256, 0, stream>>>(Wc, WcB, (long)Dn * 2 * Dn);

  // ---------------- encoder (imag branch starts at zero) -------------------
  to_seq_kernel<<<1024, 256, 0, stream>>>(train_feat, sr, NTi, Bn, Dn, HWn);
  proj_l2n_kernel<<<LTn * Bn / 16, 256, 0, stream>>>(sr,    WenT, bk_en, per, LTn * Bn, Dn, Dn);
  proj_l2n_kernel<<<LTn * Bn / 16, 256, 0, stream>>>(zerof, WenT, bk_en, pei, LTn * Bn, Dn, Dn);
  com_attn_kernel<<<dim3((LTn + 15) / 16, Bn), 256, 0, stream>>>(
      per, pei, per, pei, sr, arr, aii, LTn, LTn, Bn, Dn, Dn);
  inorm_kernel<<<dim3(NTi, Bn), 256, 0, stream>>>(sr,  arr,     memr, HWn, Bn, Dn, scale, 1);
  inorm_kernel<<<dim3(NTi, Bn), 256, 0, stream>>>(aii, nullptr, memi, HWn, Bn, Dn, scale, 0);
  // memory key projections (shared across all 6 decoder invocations)
  proj_l2n_kernel<<<LTn * Bn / 16, 256, 0, stream>>>(memr, WcrossT, bk_cross, smr, LTn * Bn, Dn, KDn);
  proj_l2n_kernel<<<LTn * Bn / 16, 256, 0, stream>>>(memi, WcrossT, bk_cross, smi, LTn * Bn, Dn, KDn);
  pos_kernel<<<1024, 256, 0, stream>>>(train_label, posr, NTi, Bn, Dn, HWn);
  mul_kernel<<<1024, 256, 0, stream>>>(memr, posr, vmr, (long)LTn * Bn * Dn);

  // ---------------- decoder layer, per image -------------------------------
  for (int img = 0; img < NTi + NSi; ++img) {
    const float* feat = (img < NTi)
        ? train_feat + (size_t)img * Bn * Dn * HWn
        : test_feat  + (size_t)(img - NTi) * Bn * Dn * HWn;
    float* tr  = T[0];
    float* s2r = T[1]; float* s2i = T[2];
    float* t1r = T[3]; float* t1i = T[4];
    float* mkr = T[5]; float* mki = T[6];
    float* t2r = T[7]; float* t2i = T[8];
    float* c3r = T[1]; float* c3i = T[2];   // s2 dead
    float* t4r = T[5]; float* t4i = T[6];   // mk dead
    float* outr = T[9]; float* outi = T[0]; // tr dead

    to_seq_kernel<<<512, 256, 0, stream>>>(feat, tr, 1, Bn, Dn, HWn);
    // self attention (ti = 0)
    proj_l2n_kernel<<<HWn * Bn / 16, 256, 0, stream>>>(tr,    WselfT, bk_self, qsr, HWn * Bn, Dn, KDn);
    proj_l2n_kernel<<<HWn * Bn / 16, 256, 0, stream>>>(zerof, WselfT, bk_self, qsi, HWn * Bn, Dn, KDn);
    com_attn_kernel<<<dim3((HWn + 15) / 16, Bn), 256, 0, stream>>>(
        qsr, qsi, qsr, qsi, tr, s2r, s2i, HWn, HWn, Bn, KDn, Dn);
    inorm_kernel<<<dim3(1, Bn), 256, 0, stream>>>(tr,  s2r,     t1r, HWn, Bn, Dn, scale, 1);
    inorm_kernel<<<dim3(1, Bn), 256, 0, stream>>>(s2i, nullptr, t1i, HWn, Bn, Dn, scale, 0);
    // cross attention queries
    proj_l2n_kernel<<<HWn * Bn / 16, 256, 0, stream>>>(t1r, WcrossT, bk_cross, pqr, HWn * Bn, Dn, KDn);
    proj_l2n_kernel<<<HWn * Bn / 16, 256, 0, stream>>>(t1i, WcrossT, bk_cross, pqi, HWn * Bn, Dn, KDn);
    // cross-1: V = positional map
    com_attn_kernel<<<dim3((HWn + 15) / 16, Bn), 256, 0, stream>>>(
        pqr, pqi, smr, smi, posr, mkr, mki, HWn, LTn, Bn, KDn, Dn);
    inorm_kernel<<<dim3(1, Bn), 256, 0, stream>>>(t1r, mkr, t2r, HWn, Bn, Dn, scale, 2);
    inorm_kernel<<<dim3(1, Bn), 256, 0, stream>>>(t1i, mki, t2i, HWn, Bn, Dn, scale, 2);
    // cross-2: V = mem * pos
    com_attn_kernel<<<dim3((HWn + 15) / 16, Bn), 256, 0, stream>>>(
        pqr, pqi, smr, smi, vmr, c3r, c3i, HWn, LTn, Bn, KDn, Dn);
    inorm_kernel<<<dim3(1, Bn), 256, 0, stream>>>(t1r, c3r, t4r, HWn, Bn, Dn, scale, 1);
    inorm_kernel<<<dim3(1, Bn), 256, 0, stream>>>(t1i, c3i, t4i, HWn, Bn, Dn, scale, 1);
    inorm_kernel<<<dim3(1, Bn), 256, 0, stream>>>(t2r, t4r, outr, HWn, Bn, Dn, scale, 1);
    inorm_kernel<<<dim3(1, Bn), 256, 0, stream>>>(t2i, t4i, outi, HWn, Bn, Dn, scale, 1);
    // 1x1 compress conv -> output slice [B, D, HW]
    compress_kernel<<<dim3((HWn + 15) / 16, Dn / 128, Bn), 256, 0, stream>>>(
        outr, outi, WcB, out + (size_t)img * Bn * Dn * HWn, HWn, Bn, Dn);
  }
}
